// CriticLSTM_59803124630136
// MI455X (gfx1250) — matrix operand order repaired and checked by reading, not compiled, not run on hardware
//
#include <hip/hip_runtime.h>
#include <hip/hip_bf16.h>
#include <math.h>

typedef __bf16 bf16_t;
typedef __attribute__((ext_vector_type(16))) __bf16 bf16x16;
typedef __attribute__((ext_vector_type(8)))  __bf16 bf16x8;
typedef __attribute__((ext_vector_type(8)))  float  f32x8;
typedef __attribute__((ext_vector_type(4)))  int    i32x4;

#define HDIM   128
#define GDIM   512          // 4*H
#define DIN    111
#define KPAD   128
#define SEQL   512
#define BATCH  1024
#define NROWS  (BATCH * SEQL)   // 524288

// --- optional CDNA5 async global->LDS path (guarded; fallback compiles) ---
#if defined(__has_builtin)
#  if __has_builtin(__builtin_amdgcn_global_load_async_to_lds_b128) && \
      __has_builtin(__builtin_amdgcn_s_wait_asynccnt)
#    define HAVE_ASYNC_LDS 1
#  endif
#endif
#ifndef HAVE_ASYNC_LDS
#  define HAVE_ASYNC_LDS 0
#endif

__device__ __forceinline__ float sigmoid_f(float x) {
    return 1.0f / (1.0f + __expf(-x));
}

// ---------------------------------------------------------------
// A fragment (16x32 bf16) from a row-major LDS tile, row stride ld.
// Two contiguous 8-elem runs per lane -> 2x ds_load_b128.
// ---------------------------------------------------------------
__device__ __forceinline__ bf16x16 load_a_frag(const bf16_t* p, int row0,
                                               int k0, int ld, int lane) {
    const int m  = lane & 15;
    const int kb = (lane >> 4) * 8;
    const bf16_t* r = p + (size_t)(row0 + m) * ld + k0 + kb;
    const bf16x8 lo = *(const bf16x8*)(r);
    const bf16x8 hi = *(const bf16x8*)(r + 16);
    return __builtin_shufflevector(lo, hi, 0,1,2,3,4,5,6,7,8,9,10,11,12,13,14,15);
}

// B fragment from fragment-linear LDS: 32 contiguous bytes per lane.
__device__ __forceinline__ bf16x16 load_b_frag_sw(const bf16_t* sw,
                                                  int fragIdx, int lane) {
    const bf16_t* p = sw + ((size_t)fragIdx * 32 + lane) * 16;
    return *(const bf16x16*)p;   // 2x ds_load_b128
}

// Stage a K x N weight matrix (row-major src W[n][ksrc], zero-pad K) into
// fragment-linear LDS: sw[((kt*NF + nt)*32 + lane)*16 + e]
__device__ __forceinline__ void stage_b_swizzled(bf16_t* sw, const float* W,
                                                 int Ksrc, int Kpad, int N,
                                                 int nBase, int NF, int tid) {
    const int total = Kpad * N;
    for (int i = tid; i < total; i += 256) {
        const int fragIdx = i >> 9;
        const int j       = i & 511;
        const int l       = j >> 4;
        const int e       = j & 15;
        const int kt      = fragIdx / NF;
        const int nt      = fragIdx % NF;
        const int k       = kt * 32 + ((l >> 4) * 8) + (e & 7) + ((e >> 3) * 16);
        const int n       = nBase + nt * 16 + (l & 15);
        const float v     = (k < Ksrc) ? W[(size_t)n * Ksrc + k] : 0.f;
        sw[i] = (bf16_t)v;
    }
}

// ====================================================================
// Kernel 1: gx = concat(obs,act) @ W_ih^T + (b_ih + b_hh)
// Time-major rows (s*BATCH + b), fragment-linear bf16 output.
// ====================================================================
__global__ __launch_bounds__(256) void gx_kernel(
    const float* __restrict__ obs, const float* __restrict__ act,
    const float* __restrict__ W_ih, const float* __restrict__ b_ih,
    const float* __restrict__ b_hh, bf16_t* __restrict__ gxf)
{
    __shared__ bf16_t xa[128][KPAD];      // 32 KB row-major activations
    __shared__ bf16_t wb_sw[16 * 512];    // 16 KB: 4 kt x 4 nt fragments

    const int tid   = threadIdx.x;
    const int lane  = tid & 31;
    const int wave  = tid >> 5;
    const int mBase = blockIdx.x * 128;   // time-major row base (fixed s)
    const int nBase = blockIdx.y * 64;

    for (int i = tid; i < 128 * KPAD; i += 256) {
        const int r = i >> 7, d = i & 127;
        const int rowp = mBase + r;
        const int b = rowp & (BATCH - 1);
        const int s = rowp >> 10;               // BATCH = 1024
        const size_t m = (size_t)b * SEQL + s;
        float v = 0.f;
        if (d < 99)       v = obs[m * 99 + d];
        else if (d < DIN) v = act[m * 12 + (d - 99)];
        xa[r][d] = (bf16_t)v;
    }
    stage_b_swizzled(wb_sw, W_ih, DIN, KPAD, 64, nBase, 4, tid);
    __syncthreads();

    f32x8 acc[4];
#pragma unroll
    for (int f = 0; f < 4; ++f) {
        const int col = nBase + f * 16 + (lane & 15);
        const float bias = b_ih[col] + b_hh[col];
#pragma unroll
        for (int v = 0; v < 8; ++v) acc[f][v] = bias;
    }

    const int rowW = wave * 16;
#pragma unroll
    for (int ks = 0; ks < 4; ++ks) {
        const bf16x16 a = load_a_frag(&xa[0][0], rowW, ks * 32, KPAD, lane);
#pragma unroll
        for (int f = 0; f < 4; ++f) {
            const bf16x16 b = load_b_frag_sw(wb_sw, ks * 4 + f, lane);
            acc[f] = __builtin_amdgcn_wmma_f32_16x16x32_bf16(
                false, a, false, b, (short)0, acc[f], false, false);
        }
    }

    const int rt = (mBase >> 4) + wave;
#pragma unroll
    for (int f = 0; f < 4; ++f) {
        const int ntg = (nBase >> 4) + f;
        bf16x8 pk;
#pragma unroll
        for (int v = 0; v < 8; ++v) pk[v] = (bf16_t)acc[f][v];
        *(bf16x8*)(gxf + ((size_t)((size_t)rt * 32 + ntg) * 32 + lane) * 8) = pk;
    }
}

// ====================================================================
// Kernel 2: LSTM recurrence. W_hh resident in LDS; wave w owns gate
// columns {g*128 + w*16} so the cell update is pure-register math and
// c stays in VGPRs for all 512 steps. gx step blocks are double-
// buffered into LDS with CDNA5 async copies when available.
// ====================================================================
__global__ __launch_bounds__(256) void lstm_kernel(
    const float* __restrict__ W_hh, const bf16_t* __restrict__ gxf,
    bf16_t* __restrict__ hs, float* __restrict__ out)
{
    extern __shared__ char smem[];
    bf16_t* whh = (bf16_t*)smem;                      // frag-linear, 128 KB
    bf16_t* ht  = (bf16_t*)(smem + HDIM * GDIM * 2);  // [16][128] bf16, 4 KB
#if HAVE_ASYNC_LDS
    bf16_t* gbuf = (bf16_t*)(smem + HDIM * GDIM * 2 + 16 * HDIM * 2); // 2 x 16 KB
#endif

    const int tid  = threadIdx.x;
    const int lane = tid & 31;
    const int wave = tid >> 5;
    const int bB   = blockIdx.x * 16;

    const size_t HT_OFF = (size_t)NROWS;
    const size_t CT_OFF = (size_t)NROWS + (size_t)BATCH * HDIM;

    stage_b_swizzled(whh, W_hh, HDIM, HDIM, GDIM, 0, 32, tid);
    for (int i = tid; i < 16 * HDIM; i += 256) ht[i] = (bf16_t)0.f;
    __syncthreads();

    const int hcol = wave * 16 + (lane & 15);
    const int half = (lane >> 4) * 8;
    float cpriv[8];
#pragma unroll
    for (int v = 0; v < 8; ++v) cpriv[v] = 0.f;

    const int rtBase = (bB >> 4);
    const int STEP_ELEMS = 32 * 32 * 8;   // 8192 bf16 = 16 KB per step block

#if HAVE_ASYNC_LDS
    // async-stage one 16 KB gx step block: 4 x (256 lanes x 16 B)
    auto issue_block = [&](int s) {
        const size_t goff = ((size_t)s * (BATCH / 16) + rtBase) * STEP_ELEMS;
        bf16_t* dst = gbuf + (size_t)(s & 1) * STEP_ELEMS;
#pragma unroll
        for (int i = 0; i < 4; ++i) {
            const size_t e = (size_t)(i * 256 + tid) * 8;   // 16 B per lane
            __builtin_amdgcn_global_load_async_to_lds_b128(
                (i32x4*)(gxf + goff + e), (i32x4*)(dst + e), 0, 0);
        }
    };
    issue_block(0);
#endif

    for (int s = 0; s < SEQL; ++s) {
#if HAVE_ASYNC_LDS
        if (s + 1 < SEQL) {
            issue_block(s + 1);                       // overlap with this step
            __builtin_amdgcn_s_wait_asynccnt(4);      // step-s block landed
        } else {
            __builtin_amdgcn_s_wait_asynccnt(0);
        }
        __syncthreads();                              // cross-wave visibility
        const bf16_t* gsrc = gbuf + (size_t)(s & 1) * STEP_ELEMS;
#else
        const size_t rt = (size_t)s * (BATCH / 16) + rtBase;
        if (s + 1 < SEQL) {
            const size_t nb = ((size_t)(s + 1) * (BATCH / 16) + rtBase)
                              * (size_t)STEP_ELEMS;
            __builtin_prefetch(gxf + nb + (size_t)tid * 32, 0, 1);
        }
        const bf16_t* gsrc = gxf + rt * STEP_ELEMS;
#endif

        // C fragments: one b128 load each; nt = gate*8 + wave
        f32x8 acc[4];
#pragma unroll
        for (int g = 0; g < 4; ++g) {
            const bf16x8 gv = *(const bf16x8*)
                (gsrc + ((size_t)(g * 8 + wave) * 32 + lane) * 8);
#pragma unroll
            for (int v = 0; v < 8; ++v) acc[g][v] = (float)gv[v];
        }
        // g += h @ W_hh^T
#pragma unroll
        for (int ks = 0; ks < 4; ++ks) {
            const bf16x16 a = load_a_frag(ht, 0, ks * 32, HDIM, lane);
#pragma unroll
            for (int g = 0; g < 4; ++g) {
                const bf16x16 b = load_b_frag_sw(whh, ks * 32 + g * 8 + wave, lane);
                acc[g] = __builtin_amdgcn_wmma_f32_16x16x32_bf16(
                    false, a, false, b, (short)0, acc[g], false, false);
            }
        }
        __syncthreads();   // all waves done reading ht of step s-1

        // LSTM cell in registers: acc[0]=i, [1]=f, [2]=g, [3]=o
#pragma unroll
        for (int v = 0; v < 8; ++v) {
            const int m = half + v;
            const float c = sigmoid_f(acc[1][v]) * cpriv[v]
                          + sigmoid_f(acc[0][v]) * tanhf(acc[2][v]);
            const float h = sigmoid_f(acc[3][v]) * tanhf(c);
            cpriv[v] = c;
            const bf16_t hb = (bf16_t)h;
            ht[m * HDIM + hcol] = hb;
            hs[((size_t)(bB + m) * SEQL + s) * HDIM + hcol] = hb;
            if (s == SEQL - 1) {
                out[HT_OFF + (size_t)(bB + m) * HDIM + hcol] = h;
                out[CT_OFF + (size_t)(bB + m) * HDIM + hcol] = c;
            }
        }
        __syncthreads();   // ht of step s visible before next WMMAs
    }
}

// ====================================================================
// Kernel 3: fused MLP head  relu(hs@W1^T+b1) -> relu(@W2^T+b2) -> @W3+b3
// ====================================================================
__global__ __launch_bounds__(256) void mlp_kernel(
    const bf16_t* __restrict__ hs,
    const float* __restrict__ W1, const float* __restrict__ b1,
    const float* __restrict__ W2, const float* __restrict__ b2,
    const float* __restrict__ W3, const float* __restrict__ b3,
    float* __restrict__ out)
{
    extern __shared__ char smem[];
    bf16_t* xin  = (bf16_t*)smem;                                       // 32 KB
    bf16_t* wsw  = (bf16_t*)(smem + 128 * 128 * 2);                     // 64 KB
    bf16_t* act1 = (bf16_t*)(smem + 128 * 128 * 2 + 128 * 256 * 2);     // 64 KB
    bf16_t* act2 = (bf16_t*)(smem + 128 * 128 * 2 + 2 * 128 * 256 * 2); // 32 KB

    const int tid   = threadIdx.x;
    const int lane  = tid & 31;
    const int wave  = tid >> 5;
    const int rBase = blockIdx.x * 128;
    const int rw    = wave * 16;

    {
        const bf16x8* src = (const bf16x8*)(hs + (size_t)rBase * HDIM);
        bf16x8* dst = (bf16x8*)xin;
        for (int i = tid; i < (128 * HDIM) / 8; i += 256) dst[i] = src[i];
    }
    stage_b_swizzled(wsw, W1, HDIM, HDIM, 256, 0, 16, tid);
    __syncthreads();

    // ---- layer 1: [128x128] x [128x256], relu ----
    bf16x16 a1[4];
#pragma unroll
    for (int ks = 0; ks < 4; ++ks) a1[ks] = load_a_frag(xin, rw, ks * 32, HDIM, lane);

#pragma unroll
    for (int f = 0; f < 16; ++f) {
        const int col = f * 16 + (lane & 15);
        f32x8 acc;
#pragma unroll
        for (int v = 0; v < 8; ++v) acc[v] = b1[col];
#pragma unroll
        for (int ks = 0; ks < 4; ++ks) {
            const bf16x16 b = load_b_frag_sw(wsw, ks * 16 + f, lane);
            acc = __builtin_amdgcn_wmma_f32_16x16x32_bf16(
                false, a1[ks], false, b, (short)0, acc, false, false);
        }
#pragma unroll
        for (int v = 0; v < 8; ++v) {
            const int m = rw + v + ((lane >> 4) * 8);
            act1[m * 256 + col] = (bf16_t)fmaxf(acc[v], 0.f);
        }
    }
    __syncthreads();

    stage_b_swizzled(wsw, W2, 256, 256, HDIM, 0, 8, tid);
    __syncthreads();

    // ---- layer 2: [128x256] x [256x128], relu ----
#pragma unroll
    for (int f = 0; f < 8; ++f) {
        const int col = f * 16 + (lane & 15);
        f32x8 acc;
#pragma unroll
        for (int v = 0; v < 8; ++v) acc[v] = b2[col];
#pragma unroll
        for (int ks = 0; ks < 8; ++ks) {
            const bf16x16 a = load_a_frag(act1, rw, ks * 32, 256, lane);
            const bf16x16 b = load_b_frag_sw(wsw, ks * 8 + f, lane);
            acc = __builtin_amdgcn_wmma_f32_16x16x32_bf16(
                false, a, false, b, (short)0, acc, false, false);
        }
#pragma unroll
        for (int v = 0; v < 8; ++v) {
            const int m = rw + v + ((lane >> 4) * 8);
            act2[m * 128 + col] = (bf16_t)fmaxf(acc[v], 0.f);
        }
    }
    __syncthreads();

    // ---- layer 3: 128-wide dot per row ----
    if (tid < 128) {
        float s = 0.f;
#pragma unroll 8
        for (int k = 0; k < 128; ++k)
            s += (float)act2[tid * 128 + k] * W3[k];
        out[(size_t)rBase + tid] = s + b3[0];
    }
}

// ====================================================================
extern "C" void kernel_launch(void* const* d_in, const int* in_sizes, int n_in,
                              void* d_out, int out_size, void* d_ws, size_t ws_size,
                              hipStream_t stream) {
    const float* obs  = (const float*)d_in[0];
    const float* act  = (const float*)d_in[1];
    const float* W_ih = (const float*)d_in[2];
    const float* W_hh = (const float*)d_in[3];
    const float* b_ih = (const float*)d_in[4];
    const float* b_hh = (const float*)d_in[5];
    const float* W1   = (const float*)d_in[6];
    const float* b1   = (const float*)d_in[7];
    const float* W2   = (const float*)d_in[8];
    const float* b2   = (const float*)d_in[9];
    const float* W3   = (const float*)d_in[10];
    const float* b3   = (const float*)d_in[11];
    float* out = (float*)d_out;

    bf16_t* gxf = (bf16_t*)d_ws;
    bf16_t* hs  = (bf16_t*)((char*)d_ws + (size_t)NROWS * GDIM * sizeof(bf16_t));

    int lstm_smem = HDIM * GDIM * 2 + 16 * HDIM * 2;        // 135168 B
#if HAVE_ASYNC_LDS
    lstm_smem += 2 * 32 * 32 * 8 * 2;                       // + 32 KB ping-pong
#endif
    const int MLP_SMEM = 128 * 128 * 2 + 2 * 128 * 256 * 2 + 128 * 128 * 2; // 196608 B
    (void)hipFuncSetAttribute((const void*)lstm_kernel,
                              hipFuncAttributeMaxDynamicSharedMemorySize, lstm_smem);
    (void)hipFuncSetAttribute((const void*)mlp_kernel,
                              hipFuncAttributeMaxDynamicSharedMemorySize, MLP_SMEM);

    gx_kernel<<<dim3(NROWS / 128, GDIM / 64), 256, 0, stream>>>(
        obs, act, W_ih, b_ih, b_hh, gxf);
    lstm_kernel<<<dim3(BATCH / 16), 256, lstm_smem, stream>>>(
        W_hh, gxf, hs, out);
    mlp_kernel<<<dim3(NROWS / 128), 256, MLP_SMEM, stream>>>(
        hs, W1, b1, W2, b2, W3, b3, out);
}
